// MultiHeadAttention_72885595013338
// MI455X (gfx1250) — matrix operand order repaired and checked by reading, not compiled, not run on hardware
//
#include <hip/hip_runtime.h>

#define NN 20000
#define EE 640000
#define HH 8
#define CC 128

typedef __attribute__((ext_vector_type(16))) _Float16 v16h;
typedef __attribute__((ext_vector_type(8)))  _Float16 v8h;
typedef __attribute__((ext_vector_type(8)))  float    v8f;

// fragment table ids (each fragment = 32 lanes x 16 halfs = 1KB)
#define F_WK1_0        24
#define F_WK1_1        25
#define F_WK2          26
#define F_WKL          27
#define F_WV1_0        28
#define F_WV1_1        29
#define F_WV2          30
#define F_WVL          31
#define F_CEN(nt, kk)  (32 + (nt) * 4 + (kk))
#define F_OUT(nt, kk)  (64 + (nt) * 4 + (kk))
#define NFRAG          96

__device__ __forceinline__ v8f wmma16(v16h a, v16h b, v8f c) {
  return __builtin_amdgcn_wmma_f32_16x16x32_f16(false, a, false, b, (short)0, c, false, false);
}

__device__ __forceinline__ float sspf(float x) {
  float sp = (x > 15.0f) ? x : log1pf(__expf(x));
  return sp - 0.69314718055994531f;
}

// ---- async Global -> LDS tile copy (GLOBAL_LOAD_ASYNC_TO_LDS_B128, ASYNCcnt) ----
template <int BYTES>
__device__ __forceinline__ void async_tile_load(void* lds_dst, const void* gsrc) {
  int lane = threadIdx.x & 31;
  unsigned ldsa = (unsigned)(uintptr_t)lds_dst + (unsigned)lane * 16u;
  const char* g = (const char*)gsrc + (size_t)lane * 16;
#pragma unroll
  for (int off = 0; off < BYTES; off += 512) {  // 32 lanes x 16B per op
    asm volatile("global_load_async_to_lds_b128 %0, %1, off offset:%2"
                 :: "v"(ldsa), "v"(g), "i"(off) : "memory");
  }
#if __has_builtin(__builtin_amdgcn_s_wait_asynccnt)
  __builtin_amdgcn_s_wait_asynccnt(0);
#else
  asm volatile("s_wait_asynccnt 0x0" ::: "memory");
#endif
}

// A fragment: full 16x32 window from f16 LDS (row-major, stride ld halfs)
__device__ __forceinline__ v16h load_a_full(const _Float16* sm, int ld) {
  int lane = threadIdx.x & 31, m = lane & 15, hi = lane >> 4;
  v16h a;
#pragma unroll
  for (int j = 0; j < 8; ++j) {
    a[j]     = sm[m * ld + hi * 8 + j];
    a[j + 8] = sm[m * ld + 16 + hi * 8 + j];
  }
  return a;
}

// A fragment: full 16x32 window from f32 LDS, cvt on read
__device__ __forceinline__ v16h load_a_full_f32(const float* sm, int ld) {
  int lane = threadIdx.x & 31, m = lane & 15, hi = lane >> 4;
  v16h a;
#pragma unroll
  for (int j = 0; j < 8; ++j) {
    a[j]     = (_Float16)sm[m * ld + hi * 8 + j];
    a[j + 8] = (_Float16)sm[m * ld + 16 + hi * 8 + j];
  }
  return a;
}

// A fragment: 16x16 window from f16 LDS, K 16..31 zero
__device__ __forceinline__ v16h load_a_k16(const _Float16* sm, int ld) {
  int lane = threadIdx.x & 31, m = lane & 15, hi = lane >> 4;
  v16h a;
#pragma unroll
  for (int j = 0; j < 8; ++j) {
    a[j]     = sm[m * ld + hi * 8 + j];
    a[j + 8] = (_Float16)0.f;
  }
  return a;
}

// A fragment: 16x16 window from f32 LDS, K 16..31 zero
__device__ __forceinline__ v16h load_a_k16_f32(const float* sm, int ld) {
  int lane = threadIdx.x & 31, m = lane & 15, hi = lane >> 4;
  v16h a;
#pragma unroll
  for (int j = 0; j < 8; ++j) {
    a[j]     = (_Float16)sm[m * ld + hi * 8 + j];
    a[j + 8] = (_Float16)0.f;
  }
  return a;
}

// B fragment: one aligned 32B vector load per lane from the prepacked table
__device__ __forceinline__ v16h load_b_frag(const _Float16* __restrict__ wb, int f) {
  const _Float16* p = wb + ((size_t)f * 32 + (threadIdx.x & 31)) * 16;
  return *(const v16h*)p;
}

__device__ __forceinline__ v8f bias_c(const float* __restrict__ bias) {
  float bv = bias[threadIdx.x & 15];
  v8f c;
#pragma unroll
  for (int r = 0; r < 8; ++r) c[r] = bv;
  return c;
}

__device__ __forceinline__ void atomicMaxF(float* a, float v) {
  if (v >= 0.f) atomicMax((int*)a, __float_as_int(v));
  else          atomicMin((unsigned int*)a, __float_as_uint(v));
}

// ---------------- prepack weights into B-fragment layout ----------------
__global__ void __launch_bounds__(32)
k_prep(const float* __restrict__ kw, const float* __restrict__ qw, const float* __restrict__ vw,
       const float* __restrict__ wk1w, const float* __restrict__ wk2w, const float* __restrict__ wklw,
       const float* __restrict__ wv1w, const float* __restrict__ wv2w, const float* __restrict__ wvlw,
       const float* __restrict__ cenw, const float* __restrict__ outw,
       _Float16* __restrict__ wb) {
  int f = blockIdx.x;
  int lane = threadIdx.x;
  const float* W;
  int ldw, k0, kmax;
  if (f < 8)        { W = kw + f * 256;        ldw = 16;  k0 = 0;  kmax = 16; }
  else if (f < 16)  { W = qw + (f - 8) * 256;  ldw = 16;  k0 = 0;  kmax = 16; }
  else if (f < 24)  { W = vw + (f - 16) * 256; ldw = 16;  k0 = 0;  kmax = 16; }
  else if (f == F_WK1_0) { W = wk1w; ldw = 64; k0 = 0;  kmax = 64; }
  else if (f == F_WK1_1) { W = wk1w; ldw = 64; k0 = 32; kmax = 64; }
  else if (f == F_WK2)   { W = wk2w; ldw = 16; k0 = 0;  kmax = 16; }
  else if (f == F_WKL)   { W = wklw; ldw = 16; k0 = 0;  kmax = 16; }
  else if (f == F_WV1_0) { W = wv1w; ldw = 64; k0 = 0;  kmax = 64; }
  else if (f == F_WV1_1) { W = wv1w; ldw = 64; k0 = 32; kmax = 64; }
  else if (f == F_WV2)   { W = wv2w; ldw = 16; k0 = 0;  kmax = 16; }
  else if (f == F_WVL)   { W = wvlw; ldw = 16; k0 = 0;  kmax = 16; }
  else if (f < 64)  { int g = f - 32; W = cenw + (size_t)(g >> 2) * 16 * 128; ldw = 128; k0 = (g & 3) * 32; kmax = 128; }
  else              { int g = f - 64; W = outw + (size_t)(g >> 2) * 16 * 128; ldw = 128; k0 = (g & 3) * 32; kmax = 128; }
  int n = lane & 15, hi = lane >> 4;
  _Float16* dst = wb + ((size_t)f * 32 + lane) * 16;
#pragma unroll
  for (int e = 0; e < 16; ++e) {
    int k = k0 + (e < 8 ? hi * 8 + e : 16 + hi * 8 + (e - 8));
    dst[e] = (k < kmax) ? (_Float16)W[n * ldw + k] : (_Float16)0.f;
  }
}

// ---------------- init ----------------
__global__ void k_init(float* __restrict__ segmax, float* __restrict__ denom,
                       float* __restrict__ aggr) {
  int i = blockIdx.x * blockDim.x + threadIdx.x;
  if (i < NN * CC) aggr[i] = 0.f;
  if (i < NN * HH) { segmax[i] = -1e30f; denom[i] = 0.f; }
}

// ---------------- node projections ----------------
__global__ void __launch_bounds__(32)
k_node_proj(const float* __restrict__ x, const _Float16* __restrict__ wb,
            _Float16* __restrict__ hk, _Float16* __restrict__ hq, _Float16* __restrict__ hv) {
  __shared__ __align__(16) float    xs[16 * 128];
  __shared__ __align__(16) _Float16 os[3][16 * 128];
  int lane = threadIdx.x;
  size_t nb = (size_t)blockIdx.x * 16;
  async_tile_load<8192>(xs, x + nb * 128);
  __syncthreads();
  int n = lane & 15, hi = lane >> 4;
  for (int h = 0; h < HH; ++h) {
    v16h a = load_a_k16_f32(xs + h * 16, 128);
#pragma unroll
    for (int p = 0; p < 3; ++p) {
      v16h b = load_b_frag(wb, p * 8 + h);
      v8f c = {};
      c = wmma16(a, b, c);
#pragma unroll
      for (int r = 0; r < 8; ++r)
        os[p][(hi * 8 + r) * 128 + h * 16 + n] = (_Float16)c[r];
    }
  }
  __syncthreads();
  _Float16* Os[3] = {hk, hq, hv};
#pragma unroll
  for (int p = 0; p < 3; ++p)
    for (int i = lane * 8; i < 2048; i += 256)
      *(v8h*)(Os[p] + nb * 128 + i) = *(const v8h*)(&os[p][i]);
}

// ---------------- edge pipeline ----------------
__global__ void __launch_bounds__(32)
k_edge_keys(const float* __restrict__ ea, const int* __restrict__ row, const int* __restrict__ col,
            const _Float16* __restrict__ wb,
            const float* __restrict__ wk1b, const float* __restrict__ wk2b,
            const float* __restrict__ wklb,
            const float* __restrict__ wv1b, const float* __restrict__ wv2b,
            const _Float16* __restrict__ hk, const _Float16* __restrict__ hq,
            float* __restrict__ qk, _Float16* __restrict__ wvout) {
  __shared__ __align__(16) float    edf[16 * 64];
  __shared__ __align__(16) _Float16 hid[16 * 16];
  __shared__ __align__(16) _Float16 wks[16 * 16];
  __shared__ __align__(16) _Float16 stg[16 * 16];
  __shared__ int cs[16], rs[16];
  int lane = threadIdx.x;
  size_t eb = (size_t)blockIdx.x * 16;
  if (lane < 16) { cs[lane] = col[eb + lane]; rs[lane] = row[eb + lane]; }
  async_tile_load<4096>(edf, ea + eb * 64);
  if (blockIdx.x + 1 < gridDim.x)
    __builtin_prefetch(ea + (eb + 16) * 64 + (size_t)lane * 32, 0, 1);
  __syncthreads();
  int n = lane & 15, hi = lane >> 4;

  // W_k = ssp(ea @ wk1^T + b) @ wk2^T + b
  v8f acc = bias_c(wk1b);
  acc = wmma16(load_a_full_f32(edf, 64),      load_b_frag(wb, F_WK1_0), acc);
  acc = wmma16(load_a_full_f32(edf + 32, 64), load_b_frag(wb, F_WK1_1), acc);
#pragma unroll
  for (int r = 0; r < 8; ++r) hid[(hi * 8 + r) * 16 + n] = (_Float16)sspf(acc[r]);
  __syncthreads();
  v8f wkacc = bias_c(wk2b);
  wkacc = wmma16(load_a_k16(hid, 16), load_b_frag(wb, F_WK2), wkacc);
#pragma unroll
  for (int r = 0; r < 8; ++r) wks[(hi * 8 + r) * 16 + n] = (_Float16)wkacc[r];
  __syncthreads();

  // W_v = ssp(ea @ wv1^T + b) @ wv2^T + b  -> staged, coalesced store
  v8f accv = bias_c(wv1b);
  accv = wmma16(load_a_full_f32(edf, 64),      load_b_frag(wb, F_WV1_0), accv);
  accv = wmma16(load_a_full_f32(edf + 32, 64), load_b_frag(wb, F_WV1_1), accv);
#pragma unroll
  for (int r = 0; r < 8; ++r) hid[(hi * 8 + r) * 16 + n] = (_Float16)sspf(accv[r]);
  __syncthreads();
  v8f wvacc = bias_c(wv2b);
  wvacc = wmma16(load_a_k16(hid, 16), load_b_frag(wb, F_WV2), wvacc);
#pragma unroll
  for (int r = 0; r < 8; ++r) stg[(hi * 8 + r) * 16 + n] = (_Float16)wvacc[r];
  __syncthreads();
  *(v8h*)(wvout + eb * 16 + lane * 8) = *(const v8h*)(&stg[lane * 8]);

  // keys_j per (edge,head) rows; qk via half-wave reduction
  v16h bwkl = load_b_frag(wb, F_WKL);
#pragma unroll 1
  for (int t = 0; t < 8; ++t) {
    int e0 = 2 * t;
    int m = lane & 15;
    int el = e0 + (m >> 3), h = m & 7;
    int cn = cs[el];
    v16h a;
    const _Float16* hp = hk + (size_t)cn * 128 + h * 16 + hi * 8;
#pragma unroll
    for (int j = 0; j < 8; ++j) {
      a[j]     = hp[j] * wks[el * 16 + hi * 8 + j];
      a[j + 8] = (_Float16)0.f;
    }
    v8f d = bias_c(wklb);
    d = wmma16(a, bwkl, d);
    float part[8];
#pragma unroll
    for (int r = 0; r < 8; ++r) {
      int mr = hi * 8 + r;
      int er = e0 + (mr >> 3), hr = mr & 7;
      part[r] = (float)hq[(size_t)rs[er] * 128 + hr * 16 + n] * d[r];
    }
#pragma unroll
    for (int r = 0; r < 8; ++r) {
      part[r] += __shfl_xor(part[r], 1);
      part[r] += __shfl_xor(part[r], 2);
      part[r] += __shfl_xor(part[r], 4);
      part[r] += __shfl_xor(part[r], 8);
    }
    if (n == 0) {
#pragma unroll
      for (int r = 0; r < 8; ++r) {
        int mr = hi * 8 + r;
        int er = e0 + (mr >> 3), hr = mr & 7;
        qk[(eb + er) * 8 + hr] = part[r] * 0.25f;
      }
    }
  }
}

// ---------------- scatter softmax ----------------
__global__ void k_segmax(const float* __restrict__ qk, const int* __restrict__ row,
                         float* __restrict__ sm) {
  int i = blockIdx.x * blockDim.x + threadIdx.x;
  if (i >= EE * HH) return;
  int e = i >> 3, h = i & 7;
  atomicMaxF(&sm[row[e] * 8 + h], qk[i]);
}

__global__ void k_expsum(const float* __restrict__ qk, const int* __restrict__ row,
                         const float* __restrict__ sm, float* __restrict__ ex,
                         float* __restrict__ denom) {
  int i = blockIdx.x * blockDim.x + threadIdx.x;
  if (i >= EE * HH) return;
  int e = i >> 3, h = i & 7;
  float v = __expf(qk[i] - sm[row[e] * 8 + h]);
  ex[i] = v;
  atomicAdd(&denom[row[e] * 8 + h], v);
}

__global__ void k_invd(float* __restrict__ denom) {
  int i = blockIdx.x * blockDim.x + threadIdx.x;
  if (i < NN * HH) denom[i] = 1.0f / denom[i];
}

// ---------------- aggregation ----------------
__global__ void __launch_bounds__(32)
k_aggr(const int* __restrict__ row, const int* __restrict__ col,
       const _Float16* __restrict__ wv, const float* __restrict__ ex,
       const float* __restrict__ rdenom, const _Float16* __restrict__ hv,
       const _Float16* __restrict__ wb, const float* __restrict__ wvlb,
       float* __restrict__ aggr) {
  __shared__ __align__(16) _Float16 wvs[16 * 16];
  __shared__ int cs[16], rs[16];
  int lane = threadIdx.x;
  size_t eb = (size_t)blockIdx.x * 16;
  if (lane < 16) { cs[lane] = col[eb + lane]; rs[lane] = row[eb + lane]; }
  *(v8h*)(&wvs[lane * 8]) = *(const v8h*)(wv + eb * 16 + lane * 8);
  __syncthreads();
  int n = lane & 15, hi = lane >> 4;
  v16h bwvl = load_b_frag(wb, F_WVL);
#pragma unroll 1
  for (int t = 0; t < 8; ++t) {
    int e0 = 2 * t;
    int m = lane & 15;
    int el = e0 + (m >> 3), h = m & 7;
    int cn = cs[el];
    v16h a;
    const _Float16* hp = hv + (size_t)cn * 128 + h * 16 + hi * 8;
#pragma unroll
    for (int j = 0; j < 8; ++j) {
      a[j]     = hp[j] * wvs[el * 16 + hi * 8 + j];
      a[j + 8] = (_Float16)0.f;
    }
    v8f d = bias_c(wvlb);
    d = wmma16(a, bwvl, d);
#pragma unroll
    for (int r = 0; r < 8; ++r) {
      int mr = hi * 8 + r;
      int er = e0 + (mr >> 3), hr = mr & 7;
      int dst = rs[er];
      float alpha = ex[(eb + er) * 8 + hr] * rdenom[dst * 8 + hr];
      atomicAdd(&aggr[(size_t)dst * 128 + hr * 16 + n], alpha * d[r]);
    }
  }
}

// ---------------- output ----------------
__global__ void __launch_bounds__(32)
k_out(const float* __restrict__ x, const float* __restrict__ aggr,
      const _Float16* __restrict__ wb,
      const float* __restrict__ cenb, const float* __restrict__ outb,
      const float* __restrict__ lng, const float* __restrict__ lnb,
      float* __restrict__ out) {
  __shared__ __align__(16) float    xs[16 * 128];
  __shared__ __align__(16) _Float16 ts[16 * 128];
  __shared__ __align__(16) float    os[16 * 128];
  __shared__ float mu_s[16], rs_s[16];
  int lane = threadIdx.x;
  size_t nb = (size_t)blockIdx.x * 16;
  async_tile_load<8192>(xs, x + nb * 128);
  __syncthreads();
  int n = lane & 15, hi = lane >> 4;
#pragma unroll 1
  for (int nt = 0; nt < 8; ++nt) {
    int n0 = nt * 16;
    v8f c;
#pragma unroll
    for (int r = 0; r < 8; ++r) {
      int m = hi * 8 + r;
      c[r] = cenb[n0 + n] + aggr[(nb + m) * 128 + n0 + n];
    }
#pragma unroll
    for (int kk = 0; kk < 4; ++kk)
      c = wmma16(load_a_full_f32(xs + kk * 32, 128), load_b_frag(wb, F_CEN(nt, kk)), c);
#pragma unroll
    for (int r = 0; r < 8; ++r) ts[(hi * 8 + r) * 128 + n0 + n] = (_Float16)sspf(c[r]);
  }
  __syncthreads();
#pragma unroll 1
  for (int nt = 0; nt < 8; ++nt) {
    int n0 = nt * 16;
    v8f c = bias_c(outb + n0);
#pragma unroll
    for (int kk = 0; kk < 4; ++kk)
      c = wmma16(load_a_full(ts + kk * 32, 128), load_b_frag(wb, F_OUT(nt, kk)), c);
#pragma unroll
    for (int r = 0; r < 8; ++r) os[(hi * 8 + r) * 128 + n0 + n] = c[r];
  }
  __syncthreads();
  if (lane < 16) {
    float mu = 0.f, m2 = 0.f;
    for (int j = 0; j < 128; ++j) { float v = os[lane * 128 + j]; mu += v; m2 += v * v; }
    mu *= (1.f / 128.f); m2 *= (1.f / 128.f);
    mu_s[lane] = mu;
    rs_s[lane] = rsqrtf(m2 - mu * mu + 1e-5f);
  }
  __syncthreads();
  for (int i = lane * 4; i < 2048; i += 128) {
    int rrow = i >> 7;
    float mu = mu_s[rrow], rstd = rs_s[rrow];
    float4 v4;
    v4.x = (os[i + 0] - mu) * rstd * lng[(i + 0) & 127] + lnb[(i + 0) & 127];
    v4.y = (os[i + 1] - mu) * rstd * lng[(i + 1) & 127] + lnb[(i + 1) & 127];
    v4.z = (os[i + 2] - mu) * rstd * lng[(i + 2) & 127] + lnb[(i + 2) & 127];
    v4.w = (os[i + 3] - mu) * rstd * lng[(i + 3) & 127] + lnb[(i + 3) & 127];
    *(float4*)(out + nb * 128 + i) = v4;
  }
}

extern "C" void kernel_launch(void* const* d_in, const int* in_sizes, int n_in,
                              void* d_out, int out_size, void* d_ws, size_t ws_size,
                              hipStream_t stream) {
  (void)in_sizes; (void)n_in; (void)out_size; (void)ws_size;
  const float* node_attr = (const float*)d_in[0];
  const int*   eidx      = (const int*)d_in[1];
  const float* edge_attr = (const float*)d_in[2];
  const float* kw   = (const float*)d_in[3];
  const float* qw   = (const float*)d_in[4];
  const float* vw   = (const float*)d_in[5];
  const float* wk1w = (const float*)d_in[6],  *wk1b = (const float*)d_in[7];
  const float* wk2w = (const float*)d_in[8],  *wk2b = (const float*)d_in[9];
  const float* wklw = (const float*)d_in[10], *wklb = (const float*)d_in[11];
  const float* wv1w = (const float*)d_in[12], *wv1b = (const float*)d_in[13];
  const float* wv2w = (const float*)d_in[14], *wv2b = (const float*)d_in[15];
  const float* wvlw = (const float*)d_in[16], *wvlb = (const float*)d_in[17];
  const float* cenw = (const float*)d_in[18], *cenb = (const float*)d_in[19];
  const float* outw = (const float*)d_in[20], *outb = (const float*)d_in[21];
  const float* lng  = (const float*)d_in[22], *lnb  = (const float*)d_in[23];
  float* outp = (float*)d_out;
  const int* row = eidx;
  const int* col = eidx + EE;

  char* p = (char*)d_ws;
  _Float16* wb    = (_Float16*)p; p += (size_t)NFRAG * 32 * 16 * sizeof(_Float16);
  _Float16* hk    = (_Float16*)p; p += (size_t)NN * 128 * sizeof(_Float16);
  _Float16* hq    = (_Float16*)p; p += (size_t)NN * 128 * sizeof(_Float16);
  _Float16* hv    = (_Float16*)p; p += (size_t)NN * 128 * sizeof(_Float16);
  _Float16* wvbuf = (_Float16*)p; p += (size_t)EE * 16 * sizeof(_Float16);
  float* qk = (float*)p; p += (size_t)EE * 8 * sizeof(float);
  float* ex = (float*)p; p += (size_t)EE * 8 * sizeof(float);
  float* sm = (float*)p; p += (size_t)NN * 8 * sizeof(float);
  float* dn = (float*)p; p += (size_t)NN * 8 * sizeof(float);
  float* ag = (float*)p; p += (size_t)NN * 128 * sizeof(float);

  k_prep<<<NFRAG, 32, 0, stream>>>(kw, qw, vw, wk1w, wk2w, wklw, wv1w, wv2w, wvlw,
                                   cenw, outw, wb);
  k_init<<<(NN * 128 + 255) / 256, 256, 0, stream>>>(sm, dn, ag);
  k_node_proj<<<NN / 16, 32, 0, stream>>>(node_attr, wb, hk, hq, hv);
  k_edge_keys<<<EE / 16, 32, 0, stream>>>(edge_attr, row, col, wb,
                                          wk1b, wk2b, wklb, wv1b, wv2b,
                                          hk, hq, qk, wvbuf);
  int tE = (EE * 8 + 255) / 256;
  k_segmax<<<tE, 256, 0, stream>>>(qk, row, sm);
  k_expsum<<<tE, 256, 0, stream>>>(qk, row, sm, ex, dn);
  k_invd<<<(NN * HH + 255) / 256, 256, 0, stream>>>(dn);
  k_aggr<<<EE / 16, 32, 0, stream>>>(row, col, wvbuf, ex, dn, hv, wb, wvlb, ag);
  k_out<<<NN / 16, 32, 0, stream>>>(node_attr, ag, wb, cenb, outb, lng, lnb, outp);
}
